// InputFeedRNNDecoder_73882027426456
// MI455X (gfx1250) — matrix operand order repaired and checked by reading, compiled, not run on hardware
//
#include <hip/hip_runtime.h>
#include <hip/hip_bf16.h>
#include <math.h>

// ---------------- Types for WMMA fragments ----------------
typedef __bf16 bf16_t;
typedef __attribute__((ext_vector_type(16))) __bf16 v16bf;
typedef __attribute__((ext_vector_type(8)))  __bf16 v8bf;
typedef __attribute__((ext_vector_type(8)))  float  v8f;

union V16BF { v16bf v; v8bf h[2]; };

// Problem constants
#define Tt 128
#define Bb 32
#define Ss 512
#define Hh 1024
#define NEGINF (-1e9f)
#define GSTRIDE (16 * 3072)  // elements between gate blocks in interleaved Wg

__device__ __forceinline__ float sigm(float x) { return 1.0f / (1.0f + __expf(-x)); }

// A fragment (16-bit A, 16x32): lane<16 -> K=0..7 in v0-3, K=16..23 in v4-7
//                               lane>=16 -> K=8..15,  K=24..31
__device__ __forceinline__ v16bf load_a_frag(const bf16_t* __restrict__ row, int ko) {
  V16BF u;
  u.h[0] = *(const v8bf*)(row + ko);
  u.h[1] = *(const v8bf*)(row + ko + 16);
  return u.v;
}
// B fragment (16-bit B, 32x16, column n = lane&15): lane<16 -> K=0..7,16..23, lane>=16 -> K=8..15,24..31
__device__ __forceinline__ v16bf load_b_frag(const bf16_t* __restrict__ col, int kh) {
  V16BF u;
  u.h[0] = *(const v8bf*)(col + (kh >> 1));
  u.h[1] = *(const v8bf*)(col + (kh >> 1) + 16);
  return u.v;
}

#define WMMA_BF16(a, b, c) \
  __builtin_amdgcn_wmma_f32_16x16x32_bf16(false, (a), false, (b), (short)0, (c), false, false)

// =====================================================================
// Fused gates GEMM + LSTM cell.
// gates[32,4096] = [emb_t | feed | h_prev] (3x [32,1024] bf16) @ Wgi + bias
// Wgi layout: [ntile(64)][gate(4)][row(16)][K=3072] -> one base pointer per
// lane; all 12 B loads per K-slice use small constant offsets.
// =====================================================================
__global__ void gates_lstm_kernel(const bf16_t* __restrict__ Aemb,      // [32][1024] (step slice)
                                  const bf16_t* __restrict__ feedbf,    // [32][1024]
                                  const bf16_t* __restrict__ hbf_prev,  // [32][1024]
                                  const bf16_t* __restrict__ Wgi,       // interleaved
                                  const float* __restrict__ bias,       // [4096]
                                  float* __restrict__ hf, float* __restrict__ cf,
                                  bf16_t* __restrict__ hbf_next) {
  int wave = (int)((blockIdx.x * blockDim.x + threadIdx.x) >> 5);  // 0..63 = ntile
  int lane = (int)(threadIdx.x & 31);
  if (wave >= Hh / 16) return;
  int l15 = lane & 15;
  int kh16 = lane >> 4;
  int ko = kh16 << 3;
  int kb = kh16 << 4;
  int nb = wave * 16;

  // one per-lane base; gate g row at + g*GSTRIDE (constant offsets in loop)
  const bf16_t* bbase = Wgi + ((size_t)(wave * 4) * 16 + l15) * 3072;

  v8f acc[8] = {};  // [tile*4 + gate]

#pragma unroll
  for (int seg = 0; seg < 3; ++seg) {
    const bf16_t* As = (seg == 0) ? Aemb : (seg == 1) ? feedbf : hbf_prev;
    const bf16_t* a0p = As + (size_t)l15 * Hh;
    const bf16_t* a1p = As + (size_t)(l15 + 16) * Hh;
    int kbase = seg * Hh;
#pragma unroll 2
    for (int kk = 0; kk < Hh; kk += 32) {
      v16bf a0 = load_a_frag(a0p + kk, ko);
      v16bf a1 = load_a_frag(a1p + kk, ko);
      int kofs = kbase + kk;
      v16bf b0 = load_b_frag(bbase + 0 * GSTRIDE + kofs, kb);
      acc[0] = WMMA_BF16(a0, b0, acc[0]);
      acc[4] = WMMA_BF16(a1, b0, acc[4]);
      v16bf b1 = load_b_frag(bbase + 1 * GSTRIDE + kofs, kb);
      acc[1] = WMMA_BF16(a0, b1, acc[1]);
      acc[5] = WMMA_BF16(a1, b1, acc[5]);
      v16bf b2 = load_b_frag(bbase + 2 * GSTRIDE + kofs, kb);
      acc[2] = WMMA_BF16(a0, b2, acc[2]);
      acc[6] = WMMA_BF16(a1, b2, acc[6]);
      v16bf b3 = load_b_frag(bbase + 3 * GSTRIDE + kofs, kb);
      acc[3] = WMMA_BF16(a0, b3, acc[3]);
      acc[7] = WMMA_BF16(a1, b3, acc[7]);
    }
  }

  int n = nb + l15;
  float bi = bias[n];
  float bff = bias[Hh + n];
  float bg = bias[2 * Hh + n];
  float bo = bias[3 * Hh + n];
  int mofs = kh16 * 8;  // C/D layout: lane>=16 holds rows M=8..15 of tile
#pragma unroll
  for (int tile = 0; tile < 2; ++tile) {
#pragma unroll
    for (int r = 0; r < 8; ++r) {
      int m = tile * 16 + mofs + r;
      float iv = sigm(acc[tile * 4 + 0][r] + bi);
      float fv = sigm(acc[tile * 4 + 1][r] + bff);
      float gv = tanhf(acc[tile * 4 + 2][r] + bg);
      float ov = sigm(acc[tile * 4 + 3][r] + bo);
      size_t idx = (size_t)m * Hh + n;
      float cn = fv * cf[idx] + iv * gv;
      float hn = ov * tanhf(cn);
      cf[idx] = cn;
      hf[idx] = hn;
      hbf_next[idx] = (bf16_t)hn;
    }
  }
}

// =====================================================================
// Generic M=32, N-wide GEMM over one or two [32][1024] bf16 A segments.
// C = [A0 | A1] @ Bt^T ; optional tanh; f32 and/or bf16 outputs.
// =====================================================================
__global__ void wmma_gemm32_2seg_kernel(const bf16_t* __restrict__ A0,
                                        const bf16_t* __restrict__ A1,  // may be null
                                        const bf16_t* __restrict__ Bt,  // [N][K]
                                        int K, float* __restrict__ Cf, int ldc,
                                        bf16_t* __restrict__ Cbf, int N, int act) {
  int wave = (int)((blockIdx.x * blockDim.x + threadIdx.x) >> 5);
  int lane = (int)(threadIdx.x & 31);
  int nb = wave * 16;
  if (nb >= N) return;
  int l15 = lane & 15;
  int kh16 = lane >> 4;
  int ko = kh16 << 3;
  int kb = kh16 << 4;

  const bf16_t* brow = Bt + (size_t)(nb + l15) * K;
  v8f acc0 = {};
  v8f acc1 = {};

#pragma unroll
  for (int seg = 0; seg < 2; ++seg) {
    const bf16_t* As = seg ? A1 : A0;
    if (!As) break;
    const bf16_t* a0p = As + (size_t)l15 * Hh;
    const bf16_t* a1p = As + (size_t)(l15 + 16) * Hh;
    int kbase = seg * Hh;
    for (int kk = 0; kk < Hh; kk += 32) {
      v16bf bfr = load_b_frag(brow + kbase + kk, kb);
      v16bf a0 = load_a_frag(a0p + kk, ko);
      v16bf a1 = load_a_frag(a1p + kk, ko);
      acc0 = WMMA_BF16(a0, bfr, acc0);
      acc1 = WMMA_BF16(a1, bfr, acc1);
    }
  }

  int n = nb + l15;
  int mofs = kh16 * 8;
#pragma unroll
  for (int r = 0; r < 8; ++r) {
    float v0 = acc0[r];
    float v1 = acc1[r];
    if (act) { v0 = tanhf(v0); v1 = tanhf(v1); }
    int m0 = r + mofs;
    int m1 = 16 + r + mofs;
    if (Cf)  { Cf[(size_t)m0 * ldc + n] = v0;   Cf[(size_t)m1 * ldc + n] = v1; }
    if (Cbf) { Cbf[(size_t)m0 * Hh + n] = (bf16_t)v0; Cbf[(size_t)m1 * Hh + n] = (bf16_t)v1; }
  }
}

// =====================================================================
// Fused attention: scores (masked, bf16 memory bank) + softmax + context.
// One block per batch row b; 512 threads (16 waves).
// =====================================================================
__global__ void attn_kernel(const float* __restrict__ q,       // [32][1024]
                            const bf16_t* __restrict__ membf,  // [S][B][H] bf16
                            const int* __restrict__ mlen,
                            float* __restrict__ attn_out,      // [32][512] (step slice)
                            bf16_t* __restrict__ ctxbf) {      // [32][1024]
  __shared__ float qs[Hh];
  __shared__ float ps[Ss];
  __shared__ float red[Ss];
  int b = blockIdx.x;
  int tid = threadIdx.x;
  int len = mlen[b];

  qs[tid] = q[(size_t)b * Hh + tid];
  qs[tid + 512] = q[(size_t)b * Hh + tid + 512];
  __syncthreads();

  // scores: one wave per s (wave-uniform mask branch)
  int wave = tid >> 5, lane = tid & 31;
  for (int s = wave; s < Ss; s += 16) {
    float sum;
    if (s < len) {
      const bf16_t* mr = membf + ((size_t)(s * Bb + b)) * Hh;
      sum = 0.0f;
      for (int j = lane; j < Hh; j += 32) sum += qs[j] * (float)mr[j];
      for (int off = 16; off > 0; off >>= 1) sum += __shfl_xor(sum, off, 32);
    } else {
      sum = NEGINF;
    }
    if (lane == 0) ps[s] = sum;
  }
  __syncthreads();

  // softmax over ps[0..511]
  float v = ps[tid];
  red[tid] = v;
  __syncthreads();
  for (int off = 256; off > 0; off >>= 1) {
    if (tid < off) red[tid] = fmaxf(red[tid], red[tid + off]);
    __syncthreads();
  }
  float mx = red[0];
  __syncthreads();
  float e = (tid < len) ? __expf(v - mx) : 0.0f;
  red[tid] = e;
  __syncthreads();
  for (int off = 256; off > 0; off >>= 1) {
    if (tid < off) red[tid] += red[tid + off];
    __syncthreads();
  }
  float pv = e / red[0];
  __syncthreads();
  ps[tid] = pv;
  attn_out[(size_t)b * Ss + tid] = pv;
  __syncthreads();

  // context: ctx[b,h] = sum_{s<len} p[s] * mem[s,b,h]
  for (int h = tid; h < Hh; h += 512) {
    float sum = 0.0f;
    for (int s = 0; s < len; ++s)
      sum += ps[s] * (float)membf[((size_t)(s * Bb + b)) * Hh + h];
    ctxbf[(size_t)b * Hh + h] = (bf16_t)sum;
  }
}

// ---------------- one-time prep kernels ----------------
// Gates weights: W[K][4096] -> interleaved bf16 [ntile][gate][row16][3072]
__global__ void transpose_gates_kernel(const float* __restrict__ W, bf16_t* __restrict__ Wgi,
                                       int K, int kofs) {
  long long i = (long long)blockIdx.x * blockDim.x + threadIdx.x;
  long long total = (long long)K * 4096;
  if (i >= total) return;
  int k = (int)(i / 4096);
  int n = (int)(i % 4096);
  int gate = n >> 10;
  int nloc = n & 1023;
  int ntile = nloc >> 4;
  int row = nloc & 15;
  size_t dstrow = ((size_t)ntile * 4 + gate) * 16 + row;
  Wgi[dstrow * 3072 + (size_t)(kofs + k)] = (bf16_t)W[i];
}

__global__ void transpose_to_bf16_kernel(const float* __restrict__ W, bf16_t* __restrict__ Wt,
                                         int K, int N, int ldt) {
  long long i = (long long)blockIdx.x * blockDim.x + threadIdx.x;
  long long total = (long long)K * N;
  if (i >= total) return;
  int k = (int)(i / N);
  int n = (int)(i % N);
  Wt[(size_t)n * ldt + k] = (bf16_t)W[i];
}

__global__ void convert_bf16_kernel(const float* __restrict__ X, bf16_t* __restrict__ Y,
                                    long long total) {
  long long i = (long long)blockIdx.x * blockDim.x + threadIdx.x;
  if (i >= total) return;
  Y[i] = (bf16_t)X[i];
}

// Pre-gather all needed embedding rows: Aemb[t][b][h] = emb[tgt[t,b]][h] (bf16)
__global__ void gather_emb_kernel(const int* __restrict__ tgt, const float* __restrict__ emb,
                                  bf16_t* __restrict__ Aemb) {
  long long i = (long long)blockIdx.x * blockDim.x + threadIdx.x;
  if (i >= (long long)Tt * Bb * Hh) return;
  int hh = (int)(i & 1023);
  int bt = (int)(i >> 10);
  int b = bt & 31;
  int t = bt >> 5;
  int tok = tgt[t * Bb + b];
  Aemb[i] = (bf16_t)emb[(long long)tok * Hh + hh];
}

__global__ void init_state_kernel(const float* __restrict__ h0, const float* __restrict__ c0,
                                  float* __restrict__ h, float* __restrict__ c,
                                  bf16_t* __restrict__ hbfA, bf16_t* __restrict__ feedbf) {
  int idx = blockIdx.x * blockDim.x + threadIdx.x;
  if (idx >= Bb * Hh) return;
  float hv = h0[idx];
  h[idx] = hv;
  c[idx] = c0[idx];
  hbfA[idx] = (bf16_t)hv;
  feedbf[idx] = (bf16_t)0.0f;
}

__global__ void final_copy_kernel(const float* __restrict__ h, const float* __restrict__ c,
                                  float* __restrict__ outH, float* __restrict__ outC) {
  int idx = blockIdx.x * blockDim.x + threadIdx.x;
  if (idx >= Bb * Hh) return;
  outH[idx] = h[idx];
  outC[idx] = c[idx];
}

extern "C" void kernel_launch(void* const* d_in, const int* in_sizes, int n_in,
                              void* d_out, int out_size, void* d_ws, size_t ws_size,
                              hipStream_t stream) {
  (void)in_sizes; (void)n_in; (void)out_size; (void)ws_size;
  const int*   tgt  = (const int*)d_in[0];
  const float* memB = (const float*)d_in[1];
  const int*   mlen = (const int*)d_in[2];
  const float* emb  = (const float*)d_in[3];
  const float* W_ih = (const float*)d_in[4];
  const float* W_hh = (const float*)d_in[5];
  const float* bias = (const float*)d_in[6];
  const float* W_a  = (const float*)d_in[7];
  const float* W_c  = (const float*)d_in[8];
  const float* h0   = (const float*)d_in[9];
  const float* c0   = (const float*)d_in[10];
  float* out = (float*)d_out;

  // Output layout (floats): outs[T,B,H] | attns[T,B,S] | hT[B,H] | cT[B,H]
  const size_t OUTS  = (size_t)Tt * Bb * Hh;
  const size_t ATTNS = (size_t)Tt * Bb * Ss;
  float* outs_base  = out;
  float* attns_base = out + OUTS;
  float* hT_out     = out + OUTS + ATTNS;
  float* cT_out     = hT_out + (size_t)Bb * Hh;

  // Workspace layout (256-aligned)
  char* ws = (char*)d_ws;
  size_t off = 0;
  auto alloc = [&](size_t bytes) { char* p = ws + off; off = (off + bytes + 255) & ~(size_t)255; return p; };
  bf16_t* Wgi   = (bf16_t*)alloc((size_t)4096 * 3072 * 2);      // 24 MB interleaved
  bf16_t* Wa_t  = (bf16_t*)alloc((size_t)1024 * 1024 * 2);      //  2 MB
  bf16_t* Wc_t  = (bf16_t*)alloc((size_t)1024 * 2048 * 2);      //  4 MB
  bf16_t* membf = (bf16_t*)alloc((size_t)Ss * Bb * Hh * 2);     // 32 MB
  bf16_t* Aemb  = (bf16_t*)alloc((size_t)Tt * Bb * Hh * 2);     //  8 MB
  bf16_t* hbfA  = (bf16_t*)alloc((size_t)Bb * Hh * 2);
  bf16_t* hbfB  = (bf16_t*)alloc((size_t)Bb * Hh * 2);
  bf16_t* feedbf= (bf16_t*)alloc((size_t)Bb * Hh * 2);
  bf16_t* ctxbf = (bf16_t*)alloc((size_t)Bb * Hh * 2);
  float* hbuf   = (float*)alloc((size_t)Bb * Hh * 4);
  float* cbuf   = (float*)alloc((size_t)Bb * Hh * 4);
  float* qbuf   = (float*)alloc((size_t)Bb * Hh * 4);

  const int TPB = 256;
  const int EW_GRID = (Bb * Hh + TPB - 1) / TPB;

  // ---- one-time prep ----
  {
    long long n1 = (long long)2048 * 4096;
    transpose_gates_kernel<<<(unsigned)((n1 + TPB - 1) / TPB), TPB, 0, stream>>>(W_ih, Wgi, 2048, 0);
    long long n2 = (long long)1024 * 4096;
    transpose_gates_kernel<<<(unsigned)((n2 + TPB - 1) / TPB), TPB, 0, stream>>>(W_hh, Wgi, 1024, 2048);
    long long n3 = (long long)1024 * 1024;
    transpose_to_bf16_kernel<<<(unsigned)((n3 + TPB - 1) / TPB), TPB, 0, stream>>>(W_a, Wa_t, 1024, 1024, 1024);
    long long n4 = (long long)2048 * 1024;
    transpose_to_bf16_kernel<<<(unsigned)((n4 + TPB - 1) / TPB), TPB, 0, stream>>>(W_c, Wc_t, 2048, 1024, 2048);
    long long nm = (long long)Ss * Bb * Hh;
    convert_bf16_kernel<<<(unsigned)((nm + TPB - 1) / TPB), TPB, 0, stream>>>(memB, membf, nm);
    long long ne = (long long)Tt * Bb * Hh;
    gather_emb_kernel<<<(unsigned)((ne + TPB - 1) / TPB), TPB, 0, stream>>>(tgt, emb, Aemb);
  }
  init_state_kernel<<<EW_GRID, TPB, 0, stream>>>(h0, c0, hbuf, cbuf, hbfA, feedbf);

  const int GEMM_TPB = 128;             // 4 waves per block
  const int G1_BLOCKS = (Hh / 16) / 4;  // 16 blocks, 64 waves
  const int GN_BLOCKS = (Hh / 16) / 4;  // 16 blocks for N=1024 GEMMs

  for (int t = 0; t < Tt; ++t) {
    bf16_t* h_prev = (t & 1) ? hbfB : hbfA;
    bf16_t* h_next = (t & 1) ? hbfA : hbfB;

    // 1) gates GEMM (emb|feed|h_prev) + LSTM cell fused
    gates_lstm_kernel<<<G1_BLOCKS, GEMM_TPB, 0, stream>>>(
        Aemb + (size_t)t * Bb * Hh, feedbf, h_prev, Wgi, bias, hbuf, cbuf, h_next);
    // 2) q = h @ W_a
    wmma_gemm32_2seg_kernel<<<GN_BLOCKS, GEMM_TPB, 0, stream>>>(
        h_next, (const bf16_t*)nullptr, Wa_t, 1024, qbuf, 1024,
        (bf16_t*)nullptr, 1024, 0);
    // 3) fused masked scores + softmax + context (bf16 memory bank)
    attn_kernel<<<Bb, Ss, 0, stream>>>(qbuf, membf, mlen,
                                       attns_base + (size_t)t * Bb * Ss, ctxbf);
    // 4) attn_h = tanh([ctx|h] @ W_c) -> outs[t] (f32) + feed (bf16)
    wmma_gemm32_2seg_kernel<<<GN_BLOCKS, GEMM_TPB, 0, stream>>>(
        ctxbf, h_next, Wc_t, 2048, outs_base + (size_t)t * Bb * Hh, 1024,
        feedbf, 1024, 1);
  }

  final_copy_kernel<<<EW_GRID, TPB, 0, stream>>>(hbuf, cbuf, hT_out, cT_out);
}
